// cheb_conv_with_SAt_GL_240518168584
// MI455X (gfx1250) — matrix operand order repaired and checked by reading, not compiled, not run on hardware
//
#include <hip/hip_runtime.h>

// Sizes from the reference
#define N_    32
#define T_    256
#define V_    128
#define FIN_  32
#define FOUT_ 64
#define K_    3
#define TT_   4            // t-values per workgroup

// LDS strides (in bf16 halves), chosen so every b128 chunk is 16B aligned
#define XS_  40            // x tile row stride   (rows m = t*128+u, 32 used)
#define TS_  40            // ThetaT row stride   (rows = k*64+o, 32 used)
#define YS_  136           // Y^T row stride      (rows = t*64+o, 128 used)

typedef __bf16 v16bf __attribute__((ext_vector_type(16)));
typedef float  v8f   __attribute__((ext_vector_type(8)));

struct alignas(16) U4   { unsigned int x, y, z, w; };
struct alignas(16) U4x2 { U4 lo, hi; };

// Scalar RNE convert (cold path / prep kernel only)
__device__ __forceinline__ unsigned short f2bf(float f) {
  unsigned int u = __float_as_uint(f);
  u += 0x7FFFu + ((u >> 16) & 1u);
  return (unsigned short)(u >> 16);
}
// Fast packed bf16 convert: bias round + V_PERM_B32 byte gather (3 VALU ops)
__device__ __forceinline__ unsigned int pack2f(float a, float b) {
  unsigned int ua = __float_as_uint(a) + 0x8000u;
  unsigned int ub = __float_as_uint(b) + 0x8000u;
  // result bytes (LSB..MSB) = a.b2, a.b3, b.b2, b.b3
  return __builtin_amdgcn_perm(ub, ua, 0x07060302u);
}
__device__ __forceinline__ v16bf load_frag(const unsigned short* p0,
                                           const unsigned short* p1) {
  U4x2 t;
  t.lo = *reinterpret_cast<const U4*>(p0);
  t.hi = *reinterpret_cast<const U4*>(p1);
  return __builtin_bit_cast(v16bf, t);
}

// ---------------------------------------------------------------------------
// Kernel 1: build A_k^T[n][v][u] = cheb_k[n,u,v] * SAtt[n,u,v] in bf16 (d_ws)
//   deg[n,j] = sum_i W[n,i,j];  L_t[u,v] = (u==v)*(deg[u]-1) - W[u,v]
//   cheb0 = I; cheb1 = L_t; cheb2 = 2*L_t*L_t - I   (elementwise, per ref)
// ---------------------------------------------------------------------------
__global__ __launch_bounds__(256) void prep_AT(const float* __restrict__ W,
                                               const float* __restrict__ SAtt,
                                               unsigned short* __restrict__ AT) {
  const int n = blockIdx.x;
  __shared__ float deg[V_];
  const float* Wn = W    + (size_t)n * V_ * V_;
  const float* Sn = SAtt + (size_t)n * V_ * V_;
  const int tid = threadIdx.x;
  if (tid < V_) {                        // column sums (axis=1 of [N,V,V])
    float s = 0.0f;
    for (int i = 0; i < V_; ++i) s += Wn[i * V_ + tid];
    deg[tid] = s;
  }
  __syncthreads();
  const size_t NVV = (size_t)N_ * V_ * V_;
  for (int e = tid; e < V_ * V_; e += 256) {
    int u = e >> 7, v = e & 127;
    float dg = (u == v) ? 1.0f : 0.0f;
    float lt = dg * (deg[u] - 1.0f) - Wn[e];
    float c2 = 2.0f * lt * lt - dg;
    float s  = Sn[e];
    size_t b = ((size_t)n * V_ + v) * V_ + u;   // transposed: [v][u]
    AT[b]           = f2bf(dg * s);
    AT[NVV + b]     = f2bf(lt * s);
    AT[2 * NVV + b] = f2bf(c2 * s);
  }
}

// ---------------------------------------------------------------------------
// Kernel 2: fused  Y_k = x·Θ_k  (WMMA, K'=f=32)  then  out += A_kᵀ·Y_k
// (WMMA, K'=u=128), accumulate over k in VGPRs, ReLU, store.
// One workgroup = one (n, 4-t tile); 8 waves.
// Per wave: t = wave>>1 is FIXED; stage-B tiles are (vt = (wave&1)*4+vl, ot),
// so A-frags are shared across ot and B-frags across vl -> 4x fewer loads.
// ---------------------------------------------------------------------------
__global__ __launch_bounds__(256) void cheb_main(const float* __restrict__ x,
                                                 const float* __restrict__ Theta,
                                                 const unsigned short* __restrict__ AT,
                                                 float* __restrict__ out) {
  extern __shared__ __align__(16) unsigned short smem[];
  unsigned short* xB  = smem;                         // [512][XS_] rows m=t*128+u
  unsigned short* ThB = smem + 512 * XS_;             // [192][TS_] rows k*64+o
  unsigned short* YT  = smem + 512 * XS_ + 192 * TS_; // [256][YS_] rows t*64+o

  const int tid  = threadIdx.x;
  const int wave = tid >> 5;
  const int lane = tid & 31;
  const int lr   = lane & 15;      // row/col within 16-tile
  const int grp  = lane >> 4;      // half-wave group

  const int n  = blockIdx.x >> 6;  // 64 t-blocks per n
  const int t0 = (blockIdx.x & 63) * TT_;

  // ---- load x tile (contiguous 4*128*32 f32) -> bf16 LDS ----
  const float* xt = x + (((size_t)n * T_ + t0) * V_) * FIN_;
  for (int i = tid; i < (TT_ * V_ * FIN_) / 4; i += 256) {
    const float4 v = *reinterpret_cast<const float4*>(xt + i * 4);
    int g = i * 4;
    int m = g >> 5;                // row = t*128+u
    int f = g & 31;
    unsigned int* d = reinterpret_cast<unsigned int*>(xB + m * XS_ + f);
    d[0] = pack2f(v.x, v.y);
    d[1] = pack2f(v.z, v.w);
  }
  // ---- load Theta transposed -> ThB[k*64+o][f] ----
  for (int i = tid; i < K_ * FIN_ * FOUT_; i += 256) {
    int k = i >> 11;               // / (32*64)
    int r = i & 2047;
    int f = r >> 6;
    int o = r & 63;
    ThB[(k * 64 + o) * TS_ + f] = f2bf(Theta[i]);
  }
  __syncthreads();

  v8f acc[16];
  const v8f vz = {};
#pragma unroll
  for (int i = 0; i < 16; ++i) acc[i] = vz;

  const int tB    = wave >> 1;         // stage-B t for this wave
  const int vbase = (wave & 1) * 4;    // stage-B v-tile base

  for (int k = 0; k < K_; ++k) {
    // Prefetch next k's A^T matrix into cache (global_prefetch_b8)
    if (k + 1 < K_) {
      const unsigned short* ATn = AT + ((size_t)((k + 1) * N_ + n)) * V_ * V_;
      __builtin_prefetch(ATn + (size_t)tid * 64, 0, 3);   // 128B per thread
    }

    // ---- Stage A: Y_k[(t,u), o] = x·Θ_k ; write transposed bf16 into YT ----
    v16bf bth[4];
#pragma unroll
    for (int ot = 0; ot < 4; ++ot) {   // Θ B-frags: invariant over m-tiles
      const unsigned short* br = ThB + (k * 64 + ot * 16 + lr) * TS_ + grp * 8;
      bth[ot] = load_frag(br, br + 16);
    }
#pragma unroll
    for (int ml = 0; ml < 4; ++ml) {
      const int mt = wave * 4 + ml;                        // m-tile 0..31
      const unsigned short* ar = xB + (mt * 16 + lr) * XS_;
      v16bf a = load_frag(ar + grp * 8, ar + 16 + grp * 8);
      const int t  = mt >> 3;
      const int u0 = (mt & 7) * 16 + grp * 8;
#pragma unroll
      for (int ot = 0; ot < 4; ++ot) {
        v8f c = __builtin_amdgcn_wmma_f32_16x16x32_bf16(false, a, false, bth[ot],
                                                        (short)0, vz, false, false);
        // D: M = j + 8*grp (row m), N = lr (col o) -> YT[t*64+o][u]
        unsigned int* d = reinterpret_cast<unsigned int*>(
            YT + (t * 64 + ot * 16 + lr) * YS_ + u0);
        d[0] = pack2f(c[0], c[1]); d[1] = pack2f(c[2], c[3]);
        d[2] = pack2f(c[4], c[5]); d[3] = pack2f(c[6], c[7]);
      }
    }
    __syncthreads();

    // ---- Stage B: out[t][v,o] += A_kᵀ[v,u] · Y_k[t][u,o] ----
    const unsigned short* ATk = AT + ((size_t)(k * N_ + n)) * V_ * V_;
#pragma unroll
    for (int us = 0; us < 4; ++us) {          // u = 128 -> 4 WMMA k-steps
      v16bf af[4], bf[4];
#pragma unroll
      for (int vl = 0; vl < 4; ++vl) {        // 4 A-frags, shared across ot
        const unsigned short* ar = ATk + ((vbase + vl) * 16 + lr) * V_ + us * 32;
        af[vl] = load_frag(ar + grp * 8, ar + 16 + grp * 8);
      }
#pragma unroll
      for (int ot = 0; ot < 4; ++ot) {        // 4 B-frags, shared across vl
        const unsigned short* br =
            YT + (tB * 64 + ot * 16 + lr) * YS_ + us * 32 + grp * 8;
        bf[ot] = load_frag(br, br + 16);
      }
#pragma unroll
      for (int vl = 0; vl < 4; ++vl)
#pragma unroll
        for (int ot = 0; ot < 4; ++ot)
          acc[vl * 4 + ot] = __builtin_amdgcn_wmma_f32_16x16x32_bf16(
              false, af[vl], false, bf[ot], (short)0, acc[vl * 4 + ot],
              false, false);
    }
    __syncthreads();
  }

  // ---- ReLU + store: out[n][t0+tB][v][o] ----
  float* op = out + (((size_t)n * T_ + (t0 + tB)) * V_) * FOUT_;
#pragma unroll
  for (int vl = 0; vl < 4; ++vl) {
#pragma unroll
    for (int ot = 0; ot < 4; ++ot) {
      const int oo = ot * 16 + lr;
#pragma unroll
      for (int j = 0; j < 8; ++j) {
        const int vv = (vbase + vl) * 16 + j + 8 * grp;
        float r = acc[vl * 4 + ot][j];
        op[(size_t)vv * FOUT_ + oo] = r > 0.0f ? r : 0.0f;
      }
    }
  }
}

// ---------------------------------------------------------------------------
// Host launcher. Inputs: x, spatial_attention, W, Theta (all f32).
// Workspace use: 3*32*128*128 bf16 = 3,145,728 bytes for A_k^T.
// ---------------------------------------------------------------------------
extern "C" void kernel_launch(void* const* d_in, const int* in_sizes, int n_in,
                              void* d_out, int out_size, void* d_ws, size_t ws_size,
                              hipStream_t stream) {
  (void)in_sizes; (void)n_in; (void)out_size; (void)ws_size;
  const float* x     = (const float*)d_in[0];
  const float* SAtt  = (const float*)d_in[1];
  const float* W     = (const float*)d_in[2];
  const float* Theta = (const float*)d_in[3];
  float* out = (float*)d_out;
  unsigned short* AT = (unsigned short*)d_ws;

  prep_AT<<<N_, 256, 0, stream>>>(W, SAtt, AT);

  const size_t shmem = (size_t)(512 * XS_ + 192 * TS_ + 256 * YS_) *
                       sizeof(unsigned short);           // 125,952 B < 320KB LDS
  cheb_main<<<dim3(N_ * (T_ / TT_)), 256, shmem, stream>>>(x, Theta, AT, out);
}
